// GCN_51273319579728
// MI455X (gfx1250) — matrix operand order repaired and checked by reading, not compiled, not run on hardware
//
#include <hip/hip_runtime.h>

// ---------------------------------------------------------------------------
// GCN on MI455X (gfx1250): bf16 WMMA GEMM + L2-resident edge scatter-add.
//   h = relu( D_in^-1/2 * A * (D_out^-1/2 * x) * W + b )   x3 layers
// ---------------------------------------------------------------------------

typedef __bf16 bf16;
typedef bf16  v16bf __attribute__((ext_vector_type(16)));
typedef float v8f   __attribute__((ext_vector_type(8)));

// ---------------- utility kernels ------------------------------------------

__global__ void zerof_kernel(float* __restrict__ p, size_t n) {
    size_t i = (size_t)blockIdx.x * blockDim.x + threadIdx.x;
    size_t stride = (size_t)gridDim.x * blockDim.x;
    for (; i < n; i += stride) p[i] = 0.0f;
}

__global__ void degrees_kernel(const int* __restrict__ src, const int* __restrict__ dst,
                               float* __restrict__ degS, float* __restrict__ degD, int E) {
    int i = blockIdx.x * blockDim.x + threadIdx.x;
    if (i < E) {
        atomicAdd(&degS[src[i]], 1.0f);
        atomicAdd(&degD[dst[i]], 1.0f);
    }
}

__global__ void norm_kernel(float* __restrict__ d, int n) {
    int i = blockIdx.x * blockDim.x + threadIdx.x;
    if (i < n) d[i] = rsqrtf(fmaxf(d[i], 1.0f));
}

// W (K x N, f32, row-major) -> Wt (Npad x Kpad, bf16, zero padded, K contiguous)
__global__ void convW_kernel(const float* __restrict__ W, bf16* __restrict__ Wt,
                             int K, int N, int Kpad, int Npad) {
    int idx = blockIdx.x * blockDim.x + threadIdx.x;
    int total = Npad * Kpad;
    if (idx >= total) return;
    int n = idx / Kpad;
    int k = idx - n * Kpad;
    float v = (n < N && k < K) ? W[(size_t)k * N + n] : 0.0f;
    Wt[(size_t)n * Kpad + k] = (bf16)v;
}

// ---------------- WMMA GEMM ------------------------------------------------
// D[M x Npad] = (rowscale .* A[M x K]) @ Wt^T    (Wt is Npad x Kpad bf16)
// One wave -> 16 rows x 64 cols (4 accumulators of 16x16).
// A-fragment (contiguous K): row = lane&15, kb = (lane>>4)*16,
//   elements 0..7 -> K = kb..kb+7, elements 8..15 -> K = kb+8..kb+15.
// B-fragment (contiguous K, same order as A): col = lane&15, K base = (lane>>4)*16,
//   16 consecutive K values packed 2-per-VGPR -> one contiguous 32B load.
// Host splits the K loop: branch-free main loop over Kfull (= K & ~31),
// then at most ONE predicated tail iteration (only layer 0, K=602).

__device__ __forceinline__ void loadA8_fast(const float* __restrict__ row, int k0,
                                            float s, v16bf& a, int base) {
    float2 f0 = *(const float2*)(row + k0 + 0);
    float2 f1 = *(const float2*)(row + k0 + 2);
    float2 f2 = *(const float2*)(row + k0 + 4);
    float2 f3 = *(const float2*)(row + k0 + 6);
    a[base + 0] = (bf16)(f0.x * s); a[base + 1] = (bf16)(f0.y * s);
    a[base + 2] = (bf16)(f1.x * s); a[base + 3] = (bf16)(f1.y * s);
    a[base + 4] = (bf16)(f2.x * s); a[base + 5] = (bf16)(f2.y * s);
    a[base + 6] = (bf16)(f3.x * s); a[base + 7] = (bf16)(f3.y * s);
}

__device__ __forceinline__ void loadA8_safe(const float* __restrict__ row, int k0, int K,
                                            float s, v16bf& a, int base) {
    for (int j = 0; j < 8; ++j) {
        int k = k0 + j;
        float v = (k < K) ? row[k] * s : 0.0f;
        a[base + j] = (bf16)v;
    }
}

__global__ __launch_bounds__(256)
void gemm_wmma_bf16(const float* __restrict__ A, const float* __restrict__ rowscale,
                    const bf16* __restrict__ Wt, float* __restrict__ Dm,
                    int M, int K, int Kfull, int Kpad, int ldd) {
    const int lane = threadIdx.x & 31;
    const int wave = threadIdx.x >> 5;
    const int tileM = (blockIdx.x * 8 + wave) * 16;
    if (tileM >= M) return;                  // wave-uniform exit (EXEC all-1 for WMMA)
    const int nbase = blockIdx.y * 64;

    // A fragment addressing
    const int rrow = tileM + (lane & 15);
    const int rclmp = rrow < M ? rrow : (M - 1);
    const float s = (rrow < M) ? rowscale[rrow] : 0.0f;
    const float* arow = A + (size_t)rclmp * K;
    const int kb = (lane >> 4) * 16;

    // B fragment addressing (contiguous bf16 in K)
    const int bcol  = lane & 15;
    const int kboff = (lane >> 4) * 16;
    const bf16* w0 = Wt + (size_t)(nbase +  0 + bcol) * Kpad + kboff;
    const bf16* w1 = Wt + (size_t)(nbase + 16 + bcol) * Kpad + kboff;
    const bf16* w2 = Wt + (size_t)(nbase + 32 + bcol) * Kpad + kboff;
    const bf16* w3 = Wt + (size_t)(nbase + 48 + bcol) * Kpad + kboff;

    v8f acc0 = {}, acc1 = {}, acc2 = {}, acc3 = {};

    // ---- branch-free main loop: all A accesses provably < Kfull <= K ----
    for (int kk = 0; kk < Kfull; kk += 32) {
        v16bf a;
        loadA8_fast(arow, kk + kb,      s, a, 0);
        loadA8_fast(arow, kk + kb + 8,  s, a, 8);
        v16bf b0 = *(const v16bf*)(w0 + kk);
        v16bf b1 = *(const v16bf*)(w1 + kk);
        v16bf b2 = *(const v16bf*)(w2 + kk);
        v16bf b3 = *(const v16bf*)(w3 + kk);
        acc0 = __builtin_amdgcn_wmma_f32_16x16x32_bf16(false, a, false, b0, (short)0, acc0, false, false);
        acc1 = __builtin_amdgcn_wmma_f32_16x16x32_bf16(false, a, false, b1, (short)0, acc1, false, false);
        acc2 = __builtin_amdgcn_wmma_f32_16x16x32_bf16(false, a, false, b2, (short)0, acc2, false, false);
        acc3 = __builtin_amdgcn_wmma_f32_16x16x32_bf16(false, a, false, b3, (short)0, acc3, false, false);
    }

    // ---- at most one predicated tail iteration (K not multiple of 32) ----
    for (int kk = Kfull; kk < Kpad; kk += 32) {
        v16bf a;
        loadA8_safe(arow, kk + kb,      K, s, a, 0);
        loadA8_safe(arow, kk + kb + 8,  K, s, a, 8);
        v16bf b0 = *(const v16bf*)(w0 + kk);
        v16bf b1 = *(const v16bf*)(w1 + kk);
        v16bf b2 = *(const v16bf*)(w2 + kk);
        v16bf b3 = *(const v16bf*)(w3 + kk);
        acc0 = __builtin_amdgcn_wmma_f32_16x16x32_bf16(false, a, false, b0, (short)0, acc0, false, false);
        acc1 = __builtin_amdgcn_wmma_f32_16x16x32_bf16(false, a, false, b1, (short)0, acc1, false, false);
        acc2 = __builtin_amdgcn_wmma_f32_16x16x32_bf16(false, a, false, b2, (short)0, acc2, false, false);
        acc3 = __builtin_amdgcn_wmma_f32_16x16x32_bf16(false, a, false, b3, (short)0, acc3, false, false);
    }

    // C/D layout: VGPR r -> M = tileM + (lane>>4)*8 + r, N = lane&15
    const int mout = tileM + (lane >> 4) * 8;
    const int nout = lane & 15;
    float* dbase = Dm + (size_t)mout * ldd + nbase + nout;
    for (int r = 0; r < 8; ++r) {
        if (mout + r < M) {
            dbase[(size_t)r * ldd +  0] = acc0[r];
            dbase[(size_t)r * ldd + 16] = acc1[r];
            dbase[(size_t)r * ldd + 32] = acc2[r];
            dbase[(size_t)r * ldd + 48] = acc3[r];
        }
    }
}

// ---------------- edge scatter-add -----------------------------------------
// One wave per edge; lanes stride the feature columns (coalesced 128B bursts,
// xw/agg are L2-resident at 102 MB < 192 MB global L2).

__global__ __launch_bounds__(256)
void scatter_kernel(const int* __restrict__ src, const int* __restrict__ dst,
                    const float* __restrict__ xw, float* __restrict__ agg,
                    int E, int ldd, int cols) {
    const int lane = threadIdx.x & 31;
    int gw = (blockIdx.x * blockDim.x + threadIdx.x) >> 5;
    const int nw = (gridDim.x * blockDim.x) >> 5;
    for (int e = gw; e < E; e += nw) {
        const int s = src[e];
        const int d = dst[e];
        const float* xs = xw + (size_t)s * ldd;
        float* ad = agg + (size_t)d * ldd;
        for (int c = lane; c < cols; c += 32)
            atomicAdd(&ad[c], xs[c]);
    }
}

// ---------------- epilogue: h = [relu](agg * norm_dst + b) ------------------

__global__ void finish_kernel(const float* __restrict__ agg, const float* __restrict__ normD,
                              const float* __restrict__ bias, float* __restrict__ out,
                              int M, int ldin, int cols, int ldout, int do_relu) {
    size_t idx = (size_t)blockIdx.x * blockDim.x + threadIdx.x;
    size_t total = (size_t)M * cols;
    if (idx >= total) return;
    int r = (int)(idx / cols);
    int c = (int)(idx - (size_t)r * cols);
    float v = agg[(size_t)r * ldin + c] * normD[r] + bias[c];
    if (do_relu) v = fmaxf(v, 0.0f);
    out[(size_t)r * ldout + c] = v;
}

// ---------------- host-side launcher ---------------------------------------

extern "C" void kernel_launch(void* const* d_in, const int* in_sizes, int n_in,
                              void* d_out, int out_size, void* d_ws, size_t ws_size,
                              hipStream_t stream) {
    const float* features = (const float*)d_in[0];
    const float* W0 = (const float*)d_in[1];
    const float* b0 = (const float*)d_in[2];
    const float* W1 = (const float*)d_in[3];
    const float* b1 = (const float*)d_in[4];
    const float* W2 = (const float*)d_in[5];
    const float* b2 = (const float*)d_in[6];
    const int* src = (const int*)d_in[7];
    const int* dst = (const int*)d_in[8];

    const int DIN = 602, DHID = 256, NCLS = 41;
    const int M = in_sizes[0] / DIN;      // 100000
    const int E = in_sizes[7];            // 3300000
    const int K0FULL = DIN & ~31;         // 576
    const int K0PAD  = (DIN + 31) & ~31;  // 608

    // workspace layout (256B aligned slices)
    char* ws = (char*)d_ws;
    size_t off = 0;
    auto take = [&](size_t bytes) -> char* {
        char* p = ws + off;
        off += (bytes + 255) & ~(size_t)255;
        return p;
    };
    float* normS = (float*)take(sizeof(float) * (size_t)M);
    float* normD = (float*)take(sizeof(float) * (size_t)M);
    bf16*  Wt    = (bf16*) take(sizeof(bf16) * (size_t)256 * K0PAD);  // max of the 3 layers
    float* B1    = (float*)take(sizeof(float) * (size_t)M * 256);
    float* B2    = (float*)take(sizeof(float) * (size_t)M * 256);
    if (off > ws_size) return;  // insufficient workspace; nothing safe to do

    const int TB = 256;
    const int gM = (M + TB - 1) / TB;
    const int gE = (E + TB - 1) / TB;
    const dim3 blk(TB);
    const int gemmX = ((M + 15) / 16 + 7) / 8;    // 8 waves of 16 rows per block

    // ---- degree normalization (shared by all layers) ----
    zerof_kernel<<<gM, blk, 0, stream>>>(normS, (size_t)M);
    zerof_kernel<<<gM, blk, 0, stream>>>(normD, (size_t)M);
    degrees_kernel<<<gE, blk, 0, stream>>>(src, dst, normS, normD, E);
    norm_kernel<<<gM, blk, 0, stream>>>(normS, M);
    norm_kernel<<<gM, blk, 0, stream>>>(normD, M);

    // ---- layer 0: features[602] -> 256, relu ----
    convW_kernel<<<(256 * K0PAD + TB - 1) / TB, blk, 0, stream>>>(W0, Wt, DIN, DHID, K0PAD, 256);
    gemm_wmma_bf16<<<dim3(gemmX, 4), blk, 0, stream>>>(features, normS, Wt, B1, M, DIN, K0FULL, K0PAD, 256);
    zerof_kernel<<<4096, blk, 0, stream>>>(B2, (size_t)M * 256);
    scatter_kernel<<<8192, blk, 0, stream>>>(src, dst, B1, B2, E, 256, 256);
    finish_kernel<<<(int)(((size_t)M * 256 + TB - 1) / TB), blk, 0, stream>>>(B2, normD, b0, B2, M, 256, 256, 256, 1);

    // ---- layer 1: 256 -> 256, relu ----
    convW_kernel<<<(256 * 256 + TB - 1) / TB, blk, 0, stream>>>(W1, Wt, DHID, DHID, 256, 256);
    gemm_wmma_bf16<<<dim3(gemmX, 4), blk, 0, stream>>>(B2, normS, Wt, B1, M, DHID, DHID, DHID, 256);
    zerof_kernel<<<4096, blk, 0, stream>>>(B2, (size_t)M * 256);
    scatter_kernel<<<8192, blk, 0, stream>>>(src, dst, B1, B2, E, 256, 256);
    finish_kernel<<<(int)(((size_t)M * 256 + TB - 1) / TB), blk, 0, stream>>>(B2, normD, b1, B2, M, 256, 256, 256, 1);

    // ---- layer 2: 256 -> 41 (padded to 64), no relu, out -> d_out ----
    convW_kernel<<<(64 * 256 + TB - 1) / TB, blk, 0, stream>>>(W2, Wt, DHID, NCLS, 256, 64);
    gemm_wmma_bf16<<<dim3(gemmX, 1), blk, 0, stream>>>(B2, normS, Wt, B1, M, DHID, DHID, DHID, 64);
    zerof_kernel<<<4096, blk, 0, stream>>>(B2, (size_t)M * 64);
    scatter_kernel<<<8192, blk, 0, stream>>>(src, dst, B1, B2, E, 64, NCLS);
    finish_kernel<<<(int)(((size_t)M * NCLS + TB - 1) / TB), blk, 0, stream>>>(B2, normD, b2, (float*)d_out, M, 64, NCLS, NCLS, 0);
}
